// ConditionalMoEWithLoadBalancing_48421461295382
// MI455X (gfx1250) — matrix operand order repaired and checked by reading, not compiled, not run on hardware
//
#include <hip/hip_runtime.h>
#include <math.h>

// ---------------------------------------------------------------------------
// Problem constants (from reference): x (2,2048,1024), E=8 experts,
// W1 (8,1024,2048), W2 (8,2048,1024), router D->8, difficulty head 1024->512->1
// ---------------------------------------------------------------------------
#define T_TOK 4096
#define D_IN  1024
#define H_FF  2048
#define E_NUM 8
#define DH    512

typedef __attribute__((ext_vector_type(16))) _Float16 v16h;
typedef __attribute__((ext_vector_type(8)))  _Float16 v8h;
typedef __attribute__((ext_vector_type(8)))  float    v8f;
typedef __attribute__((ext_vector_type(4)))  unsigned int u32x4;
typedef __attribute__((ext_vector_type(8)))  unsigned int u32x8;

// ---------------------------------------------------------------------------
// Workspace layout (bytes)
// ---------------------------------------------------------------------------
#define OFF_XH    ((size_t)0)                    // f16 [T, D]        8 MB
#define OFF_H1H   ((size_t)8388608)              // f16 [T, H]       16 MB
#define OFF_W1T   ((size_t)25165824)             // f16 [H, D] n-major 4 MB
#define OFF_W2T   ((size_t)29360128)             // f16 [D, H] n-major 4 MB
#define OFF_W     ((size_t)33554432)             // f32 [T, E]      128 KB
#define OFF_USAGE ((size_t)33685504)             // f32 [E]

// ---------------------------------------------------------------------------
// zero out + usage
// ---------------------------------------------------------------------------
__global__ void zero_init_kernel(float* __restrict__ out, float* __restrict__ usage) {
  size_t idx = (size_t)blockIdx.x * blockDim.x + threadIdx.x;
  size_t stride = (size_t)gridDim.x * blockDim.x;
  for (size_t i = idx; i < (size_t)T_TOK * D_IN; i += stride) out[i] = 0.0f;
  if (idx < E_NUM) usage[idx] = 0.0f;
}

// ---------------------------------------------------------------------------
// fp32 -> fp16 conversions
// ---------------------------------------------------------------------------
__global__ void convert_x_kernel(const float* __restrict__ x, _Float16* __restrict__ xh) {
  size_t idx = (size_t)blockIdx.x * blockDim.x + threadIdx.x;
  size_t stride = (size_t)gridDim.x * blockDim.x;
  for (size_t i = idx; i < (size_t)T_TOK * D_IN; i += stride)
    xh[i] = (_Float16)x[i];
}

// W1[e] is (D, H) row-major; produce n-major W1hT[n*D + k]
__global__ void convert_w1T_kernel(const float* __restrict__ w1e, _Float16* __restrict__ w1t) {
  size_t o = (size_t)blockIdx.x * blockDim.x + threadIdx.x;   // H*D elements
  if (o >= (size_t)H_FF * D_IN) return;
  int n = (int)(o >> 10);          // / D_IN
  int k = (int)(o & (D_IN - 1));
  w1t[o] = (_Float16)w1e[(size_t)k * H_FF + n];
}

// W2[e] is (H, D) row-major; produce n-major W2hT[n*H + k]
__global__ void convert_w2T_kernel(const float* __restrict__ w2e, _Float16* __restrict__ w2t) {
  size_t o = (size_t)blockIdx.x * blockDim.x + threadIdx.x;   // D*H elements
  if (o >= (size_t)D_IN * H_FF) return;
  int n = (int)(o >> 11);          // / H_FF
  int k = (int)(o & (H_FF - 1));
  w2t[o] = (_Float16)w2e[(size_t)k * D_IN + n];
}

// ---------------------------------------------------------------------------
// Router + difficulty head: one block (256 thr) per token.
// ---------------------------------------------------------------------------
__global__ __launch_bounds__(256)
void router_kernel(const float* __restrict__ x,
                   const float* __restrict__ Wd1, const float* __restrict__ bd1,
                   const float* __restrict__ Wd2, const float* __restrict__ bd2,
                   const float* __restrict__ Wg,  const float* __restrict__ bg,
                   float* __restrict__ w, float* __restrict__ usage) {
  __shared__ float sx[D_IN];
  __shared__ float sp[256];
  __shared__ float sl[256];
  __shared__ float slog[E_NUM];
  __shared__ float sent;

  const int t = blockIdx.x;
  const int tid = threadIdx.x;
  const float* xr = x + (size_t)t * D_IN;
  for (int i = tid; i < D_IN; i += 256) sx[i] = xr[i];
  __syncthreads();

  float local = 0.0f;
  #pragma unroll
  for (int jj = 0; jj < 2; ++jj) {
    int j = tid * 2 + jj;
    float s = bd1[j];
    for (int d = 0; d < D_IN; ++d) s += sx[d] * Wd1[(size_t)d * DH + j];
    s = s > 0.0f ? s : 0.0f;
    local += s * Wd2[j];
  }
  sp[tid] = local;
  __syncthreads();
  for (int off = 128; off > 0; off >>= 1) {
    if (tid < off) sp[tid] += sp[tid + off];
    __syncthreads();
  }
  if (tid == 0) {
    float z = sp[0] + bd2[0];
    sent = fmaxf(z, 0.0f) + log1pf(expf(-fabsf(z)));   // softplus
  }

  const int e = tid & 7, g = tid >> 3;
  float ls = 0.0f;
  for (int d = g; d < D_IN; d += 32) ls += sx[d] * Wg[(size_t)d * E_NUM + e];
  sl[tid] = ls;
  __syncthreads();
  if (tid < E_NUM) {
    float s = bg[tid];
    for (int gg = 0; gg < 32; ++gg) s += sl[gg * E_NUM + tid];
    slog[tid] = s;
  }
  __syncthreads();

  if (tid == 0) {
    float ne = (sent - 0.5f) * (1.0f / 1.5f);
    ne = fminf(fmaxf(ne, 0.0f), 1.0f);
    int k = (int)rintf(1.0f + ne * 7.0f);
    k = k < 1 ? 1 : (k > 8 ? 8 : k);

    float lg[E_NUM]; int ord[E_NUM];
    #pragma unroll
    for (int i = 0; i < E_NUM; ++i) { lg[i] = slog[i]; ord[i] = i; }
    for (int i = 0; i < E_NUM; ++i)
      for (int j = i + 1; j < E_NUM; ++j)
        if (lg[j] > lg[i]) {
          float tf = lg[i]; lg[i] = lg[j]; lg[j] = tf;
          int ti = ord[i]; ord[i] = ord[j]; ord[j] = ti;
        }

    float den = 0.0f, ws[E_NUM];
    float mx = lg[0];
    for (int i = 0; i < k; ++i) { ws[i] = expf(lg[i] - mx); den += ws[i]; }
    float* wr = w + (size_t)t * E_NUM;
    for (int i = 0; i < E_NUM; ++i) wr[ord[i]] = (i < k) ? (ws[i] / den) : 0.0f;

    float mx2 = slog[0];
    for (int i = 1; i < E_NUM; ++i) mx2 = fmaxf(mx2, slog[i]);
    float d2 = 0.0f, pe[E_NUM];
    for (int i = 0; i < E_NUM; ++i) { pe[i] = expf(slog[i] - mx2); d2 += pe[i]; }
    float scale = (float)k * 0.125f / d2;
    for (int i = 0; i < E_NUM; ++i) atomicAdd(&usage[i], pe[i] * scale);
  }
}

// ---------------------------------------------------------------------------
// Tensor Data Mover: issue a 2-D tile load (rows x 32 K-halves, f16) into LDS.
// D# built per CDNA5 ISA; 2-group (<=2D) form.
// ---------------------------------------------------------------------------
__device__ __forceinline__ unsigned rfl(unsigned v) {
  return (unsigned)__builtin_amdgcn_readfirstlane((int)v);
}

__device__ __forceinline__ void tdm_load_tile(const _Float16* gsrc, unsigned lds_off,
                                              unsigned tensor_d0, unsigned tensor_d1,
                                              unsigned tile_d0, unsigned tile_d1,
                                              unsigned stride0) {
  unsigned long long ga = (unsigned long long)(uintptr_t)gsrc;
  u32x4 g0;
  g0[0] = rfl(1u);                                        // count=1, user D#
  g0[1] = rfl(lds_off);                                   // lds_addr
  g0[2] = rfl((unsigned)(ga & 0xFFFFFFFFu));              // global_addr[31:0]
  g0[3] = rfl((unsigned)((ga >> 32) & 0x01FFFFFFu) | 0x80000000u); // addr[56:32] | type=2
  u32x8 g1;
  g1[0] = rfl(0x00010000u);                               // mask=0, data_size=1 (2B)
  g1[1] = rfl((tensor_d0 & 0xFFFFu) << 16);               // tensor_dim0[15:0]
  g1[2] = rfl((tensor_d0 >> 16) | ((tensor_d1 & 0xFFFFu) << 16)); // dim0 hi | dim1 lo
  g1[3] = rfl((tensor_d1 >> 16) | (tile_d0 << 16));       // dim1 hi | tile_dim0
  g1[4] = rfl(tile_d1 & 0xFFFFu);                         // tile_dim1, tile_dim2=0
  g1[5] = rfl(stride0);                                   // tensor_dim0_stride[31:0]
  g1[6] = rfl(0u);                                        // stride0 hi / dim1_stride lo
  g1[7] = rfl(0u);
  asm volatile("tensor_load_to_lds %0, %1" :: "s"(g0), "s"(g1) : "memory");
}

// ---------------------------------------------------------------------------
// WMMA fragment helpers (per CDNA5 ISA layouts, wave32)
// ---------------------------------------------------------------------------
__device__ __forceinline__ v16h lds_a_frag(const _Float16* aRow, int kb) {
  return *(const v16h*)(aRow + kb);
}

// ---------------------------------------------------------------------------
// Shared TDM-fed WMMA mainloop.
//  Block tile: 128 (M) x 128 (N), K-step 32, LDS double-buffered.
//  8 waves in 4 (M) x 2 (N) grid; each wave: 32x64 register tile = 8 WMMAs
//  per K-step (2 A frags x 4 B frags, B reused across both A rows).
//  Aglob = A + row0*KDIM (row-major, KDIM stride); Bglob = BhT + nb*KDIM.
// ---------------------------------------------------------------------------
template <int KDIM, int BROWS>
__device__ __forceinline__ void gemm_mainloop(const _Float16* __restrict__ Aglob,
                                              const _Float16* __restrict__ Bglob,
                                              _Float16 (*sA)[128 * 32],
                                              _Float16 (*sB)[128 * 32],
                                              int wave, int lane,
                                              v8f (&acc)[2][4]) {
  const int kbA   = (lane < 16) ? 0 : 16;
  const int kbB   = (lane < 16) ? 0 : 16;
  const int mwave = wave >> 1;            // 0..3  (32-row group)
  const int nwave = wave & 1;             // 0..1  (64-col group)
  const int ml    = mwave * 32 + (lane & 15);
  const int nl    = lane & 15;

  if (wave == 0) {
    tdm_load_tile(Aglob, (unsigned)(uintptr_t)&sA[0][0], KDIM, T_TOK, 32, 128, KDIM);
    tdm_load_tile(Bglob, (unsigned)(uintptr_t)&sB[0][0], KDIM, BROWS, 32, 128, KDIM);
  }

  for (int k0 = 0; k0 < KDIM; k0 += 32) {
    const int buf = (k0 >> 5) & 1;
    if (wave == 0) {
      if (k0 + 32 < KDIM) {
        const int nbuf = buf ^ 1;
        tdm_load_tile(Aglob + k0 + 32, (unsigned)(uintptr_t)&sA[nbuf][0], KDIM, T_TOK, 32, 128, KDIM);
        tdm_load_tile(Bglob + k0 + 32, (unsigned)(uintptr_t)&sB[nbuf][0], KDIM, BROWS, 32, 128, KDIM);
        __builtin_amdgcn_s_wait_tensorcnt(2);   // current buffer's 2 ops done
      } else {
        __builtin_amdgcn_s_wait_tensorcnt(0);
      }
    }
    __syncthreads();                             // LDS tile visible to all waves

    v16h a0 = lds_a_frag(&sA[buf][(ml +  0) * 32], kbA);
    v16h a1 = lds_a_frag(&sA[buf][(ml + 16) * 32], kbA);
    #pragma unroll
    for (int tn = 0; tn < 4; ++tn) {
      v16h b = *(const v16h*)(&sB[buf][(nwave * 64 + tn * 16 + nl) * 32 + kbB]);
      acc[0][tn] = __builtin_amdgcn_wmma_f32_16x16x32_f16(false, a0, false, b, (short)0, acc[0][tn], false, false);
      acc[1][tn] = __builtin_amdgcn_wmma_f32_16x16x32_f16(false, a1, false, b, (short)0, acc[1][tn], false, false);
    }
    __syncthreads();                             // done reading buf before TDM reuses it
  }
}

// ---------------------------------------------------------------------------
// GEMM1: H1 = relu(Xh(T,D) @ W1hT' + b1)   M=4096 N=2048 K=1024
// ---------------------------------------------------------------------------
__global__ __launch_bounds__(256)
void gemm1_wmma_kernel(const _Float16* __restrict__ Xh,
                       const _Float16* __restrict__ W1hT,
                       const float* __restrict__ b1e,
                       _Float16* __restrict__ H1h) {
  __shared__ _Float16 sA[2][128 * 32];
  __shared__ _Float16 sB[2][128 * 32];

  const int lane = threadIdx.x & 31;
  const int wave = threadIdx.x >> 5;
  const int row0 = blockIdx.y * 128;
  const int nb   = blockIdx.x * 128;

  v8f acc[2][4] = {};
  gemm_mainloop<D_IN, H_FF>(Xh + (size_t)row0 * D_IN, W1hT + (size_t)nb * D_IN,
                            sA, sB, wave, lane, acc);

  const int nl = lane & 15;
  const int mwave = wave >> 1, nwave = wave & 1;
  #pragma unroll
  for (int i = 0; i < 2; ++i) {
    #pragma unroll
    for (int r = 0; r < 8; ++r) {
      int m = row0 + mwave * 32 + i * 16 + ((lane < 16) ? r : (8 + r));
      _Float16* orow = H1h + (size_t)m * H_FF;
      #pragma unroll
      for (int tn = 0; tn < 4; ++tn) {
        int n = nb + nwave * 64 + tn * 16 + nl;
        float v = acc[i][tn][r] + b1e[n];
        v = v > 0.0f ? v : 0.0f;                 // relu
        orow[n] = (_Float16)v;
      }
    }
  }
}

// ---------------------------------------------------------------------------
// GEMM2: out += w[:,e] * (H1(T,H) @ W2hT' + b2)   M=4096 N=1024 K=2048
// ---------------------------------------------------------------------------
__global__ __launch_bounds__(256)
void gemm2_wmma_kernel(const _Float16* __restrict__ H1h,
                       const _Float16* __restrict__ W2hT,
                       const float* __restrict__ b2e,
                       const float* __restrict__ w,
                       int e,
                       float* __restrict__ out) {
  __shared__ _Float16 sA[2][128 * 32];
  __shared__ _Float16 sB[2][128 * 32];

  const int lane = threadIdx.x & 31;
  const int wave = threadIdx.x >> 5;
  const int row0 = blockIdx.y * 128;
  const int nb   = blockIdx.x * 128;

  v8f acc[2][4] = {};
  gemm_mainloop<H_FF, D_IN>(H1h + (size_t)row0 * H_FF, W2hT + (size_t)nb * H_FF,
                            sA, sB, wave, lane, acc);

  const int nl = lane & 15;
  const int mwave = wave >> 1, nwave = wave & 1;
  #pragma unroll
  for (int i = 0; i < 2; ++i) {
    #pragma unroll
    for (int r = 0; r < 8; ++r) {
      int m = row0 + mwave * 32 + i * 16 + ((lane < 16) ? r : (8 + r));
      float wt = w[(size_t)m * E_NUM + e];
      float* orow = out + (size_t)m * D_IN;
      #pragma unroll
      for (int tn = 0; tn < 4; ++tn) {
        int n = nb + nwave * 64 + tn * 16 + nl;
        float v = acc[i][tn][r] + b2e[n];
        orow[n] += wt * v;   // experts serialized on stream -> race-free
      }
    }
  }
}

// ---------------------------------------------------------------------------
// cv = std(usage, ddof=1) / (mean + 1e-6)
// ---------------------------------------------------------------------------
__global__ void cv_kernel(const float* __restrict__ usage, float* __restrict__ out_cv) {
  if (threadIdx.x == 0) {
    float m = 0.0f;
    for (int i = 0; i < E_NUM; ++i) m += usage[i];
    m *= 0.125f;
    float v = 0.0f;
    for (int i = 0; i < E_NUM; ++i) { float d = usage[i] - m; v += d * d; }
    v *= (1.0f / 7.0f);
    out_cv[0] = sqrtf(v) / (m + 1e-6f);
  }
}

// ---------------------------------------------------------------------------
// host launcher
// ---------------------------------------------------------------------------
extern "C" void kernel_launch(void* const* d_in, const int* in_sizes, int n_in,
                              void* d_out, int out_size, void* d_ws, size_t ws_size,
                              hipStream_t stream) {
  (void)in_sizes; (void)n_in; (void)out_size; (void)ws_size;
  const float* x   = (const float*)d_in[0];
  const float* W1  = (const float*)d_in[1];
  const float* b1  = (const float*)d_in[2];
  const float* W2  = (const float*)d_in[3];
  const float* b2  = (const float*)d_in[4];
  const float* Wg  = (const float*)d_in[5];
  const float* bg  = (const float*)d_in[6];
  const float* Wd1 = (const float*)d_in[7];
  const float* bd1 = (const float*)d_in[8];
  const float* Wd2 = (const float*)d_in[9];
  const float* bd2 = (const float*)d_in[10];
  float* out = (float*)d_out;

  char* ws = (char*)d_ws;
  _Float16* Xh    = (_Float16*)(ws + OFF_XH);
  _Float16* H1h   = (_Float16*)(ws + OFF_H1H);
  _Float16* W1hT  = (_Float16*)(ws + OFF_W1T);
  _Float16* W2hT  = (_Float16*)(ws + OFF_W2T);
  float*    wbuf  = (float*)(ws + OFF_W);
  float*    usage = (float*)(ws + OFF_USAGE);

  zero_init_kernel<<<dim3(4096), 256, 0, stream>>>(out, usage);
  convert_x_kernel<<<dim3(4096), 256, 0, stream>>>(x, Xh);
  router_kernel<<<dim3(T_TOK), 256, 0, stream>>>(x, Wd1, bd1, Wd2, bd2, Wg, bg, wbuf, usage);

  const size_t WPE = (size_t)D_IN * H_FF;   // elements per expert weight
  for (int e = 0; e < E_NUM; ++e) {
    convert_w1T_kernel<<<dim3(8192), 256, 0, stream>>>(W1 + (size_t)e * WPE, W1hT);
    gemm1_wmma_kernel<<<dim3(H_FF / 128, T_TOK / 128), 256, 0, stream>>>(
        Xh, W1hT, b1 + (size_t)e * H_FF, H1h);
    convert_w2T_kernel<<<dim3(8192), 256, 0, stream>>>(W2 + (size_t)e * WPE, W2hT);
    gemm2_wmma_kernel<<<dim3(D_IN / 128, T_TOK / 128), 256, 0, stream>>>(
        H1h, W2hT, b2 + (size_t)e * D_IN, wbuf, e, out);
  }
  cv_kernel<<<1, 32, 0, stream>>>(usage, out + (size_t)T_TOK * D_IN);
}
